// Attention_128849019066
// MI455X (gfx1250) — matrix operand order repaired and checked by reading, not compiled, not run on hardware
//
#include <hip/hip_runtime.h>

// ---------------- Types ----------------
typedef __attribute__((ext_vector_type(16))) __bf16 v16bf;
typedef __attribute__((ext_vector_type(8)))  __bf16 v8bf;
typedef __attribute__((ext_vector_type(8)))  float  v8f;

// Problem sizes
#define BB 4
#define SS 2048
#define DD 1024
#define HH 16
#define HD 64
#define MM (BB*SS)      // 8192
#define N3 (3*DD)       // 3072

// ---------------- Helpers ----------------
__device__ __forceinline__ v8f vzero8() {
  v8f z;
#pragma unroll
  for (int i = 0; i < 8; ++i) z[i] = 0.0f;
  return z;
}

__device__ __forceinline__ __bf16 f2bf(float f) {
  union { float f; unsigned u; } in; in.f = f;
  unsigned r = in.u + 0x7fffu + ((in.u >> 16) & 1u);   // RNE
  union { unsigned short s; __bf16 b; } out;
  out.s = (unsigned short)(r >> 16);
  return out.b;
}

union AFrag { v16bf v; v8bf h[2]; };

// A fragment (16x32 bf16, row-major source, stride ld):
// lane l: row = base + (l&15); elems 0..15 -> K=k0+hi*16..+15 (contiguous)
__device__ __forceinline__ v16bf load_a(const __bf16* __restrict__ base, int ld,
                                        int row, int k0, int hi) {
  AFrag a;
  const __bf16* p = base + (size_t)row * ld + k0 + hi * 16;
  a.h[0] = *(const v8bf*)(p);
  a.h[1] = *(const v8bf*)(p + 8);
  return a.v;
}

// B fragment (32x16 bf16) read from B^T stored [N][K] row-major (stride ld):
// lane l: column n = base + (l&15); K = k0 + hi*16 .. +15 (contiguous 32B)
__device__ __forceinline__ v16bf load_bt(const __bf16* __restrict__ bt, int ld,
                                         int n, int k0, int hi) {
  return *(const v16bf*)(bt + (size_t)n * ld + k0 + hi * 16);
}

__device__ __forceinline__ v8f wmma_bf16(v16bf a, v16bf b, v8f c) {
  return __builtin_amdgcn_wmma_f32_16x16x32_bf16(
      false, a, false, b, (short)0, c, false, false);
}

// ---------------- Conversion kernels ----------------
__global__ void cvt_x_kernel(const float* __restrict__ x, __bf16* __restrict__ xb) {
  int i = blockIdx.x * 256 + threadIdx.x;
  xb[i] = f2bf(x[i]);
}

// Wqkv [D][3D] -> WqkvT bf16 [3D][D]
__global__ void cvt_wqkv_kernel(const float* __restrict__ w, __bf16* __restrict__ wt) {
  int i = blockIdx.x * 256 + threadIdx.x;   // i = k*3072 + n (coalesced read)
  int k = i / N3, n = i % N3;
  wt[n * DD + k] = f2bf(w[i]);
}

// Wo [D][D] -> WoT bf16 [D][D]
__global__ void cvt_wo_kernel(const float* __restrict__ w, __bf16* __restrict__ wt) {
  int i = blockIdx.x * 256 + threadIdx.x;   // i = k*1024 + n
  int k = i >> 10, n = i & 1023;
  wt[n * DD + k] = f2bf(w[i]);
}

// ---------------- GEMM 1: qkv = xb @ Wqkv, scatter to Q / K / V^T ----------------
// grid (M/256, N3/64), block 256 (8 waves); wave tile: 32(M) x 64(N).
// Register double-buffered + fully unrolled (immediate-offset loads, no rotation movs).
__global__ __launch_bounds__(256) void gemm_qkv_kernel(
    const __bf16* __restrict__ xb, const __bf16* __restrict__ wT,
    __bf16* __restrict__ Q, __bf16* __restrict__ K, __bf16* __restrict__ V) {
  const int lane = threadIdx.x & 31, wave = threadIdx.x >> 5;
  const int nl = lane & 15, hi = lane >> 4;
  const int m0 = blockIdx.x * 256 + wave * 32;
  const int n0 = blockIdx.y * 64;

  v8f c[2][4];
#pragma unroll
  for (int s = 0; s < 2; ++s)
#pragma unroll
    for (int t = 0; t < 4; ++t) c[s][t] = vzero8();

  v16bf a_c[2], b_c[4];
#pragma unroll
  for (int s = 0; s < 2; ++s) a_c[s] = load_a(xb, DD, m0 + s * 16 + nl, 0, hi);
#pragma unroll
  for (int t = 0; t < 4; ++t) b_c[t] = load_bt(wT, DD, n0 + t * 16 + nl, 0, hi);

#pragma unroll
  for (int k0 = 0; k0 < DD - 32; k0 += 32) {
    v16bf a_n[2], b_n[4];
#pragma unroll
    for (int s = 0; s < 2; ++s) a_n[s] = load_a(xb, DD, m0 + s * 16 + nl, k0 + 32, hi);
#pragma unroll
    for (int t = 0; t < 4; ++t) b_n[t] = load_bt(wT, DD, n0 + t * 16 + nl, k0 + 32, hi);

#pragma unroll
    for (int s = 0; s < 2; ++s)
#pragma unroll
      for (int t = 0; t < 4; ++t) c[s][t] = wmma_bf16(a_c[s], b_c[t], c[s][t]);

#pragma unroll
    for (int s = 0; s < 2; ++s) a_c[s] = a_n[s];
#pragma unroll
    for (int t = 0; t < 4; ++t) b_c[t] = b_n[t];
  }
  // final k-step (no more loads)
#pragma unroll
  for (int s = 0; s < 2; ++s)
#pragma unroll
    for (int t = 0; t < 4; ++t) c[s][t] = wmma_bf16(a_c[s], b_c[t], c[s][t]);

  // Scatter epilogue; Q gets the 1/sqrt(HD) attention scale folded in.
#pragma unroll
  for (int s = 0; s < 2; ++s) {
#pragma unroll
    for (int t = 0; t < 4; ++t) {
      const int n = n0 + t * 16 + nl;
      const int sec = n >> 10, cc = n & 1023;
      const int h = cc >> 6, d = cc & 63;
#pragma unroll
      for (int r = 0; r < 8; ++r) {
        const int gm = m0 + s * 16 + r + 8 * hi;
        const int b_ = gm >> 11, sq = gm & (SS - 1);
        const int bh = b_ * HH + h;
        const float v = c[s][t][r];
        if (sec == 0)      Q[(bh * SS + sq) * HD + d] = f2bf(v * 0.125f);
        else if (sec == 1) K[(bh * SS + sq) * HD + d] = f2bf(v);
        else               V[(bh * HD + d) * SS + sq] = f2bf(v);   // transposed V
      }
    }
  }
}

// ---------------- Flash attention ----------------
// grid (S/16/8, B*H), block 256 (8 waves). Each wave: 16 queries, causal key sweep
// in 32-key tiles; K fragments double-buffered, V loads hoisted over softmax VALU.
__global__ __launch_bounds__(256) void attn_kernel(
    const __bf16* __restrict__ Q, const __bf16* __restrict__ K,
    const __bf16* __restrict__ V, __bf16* __restrict__ yb) {
  __shared__ __align__(64) __bf16 lds[8][16 * 32];   // per-wave P staging (C->A relayout)

  const int lane = threadIdx.x & 31, wave = threadIdx.x >> 5;
  const int nl = lane & 15, hi = lane >> 4;
  const int q0 = (blockIdx.x * 8 + wave) * 16;
  const int bh = blockIdx.y;
  const int b_ = bh / HH, h = bh % HH;

  const __bf16* Qp = Q + (size_t)bh * SS * HD;   // [S][64] (pre-scaled by 0.125)
  const __bf16* Kp = K + (size_t)bh * SS * HD;   // [S][64]
  const __bf16* Vp = V + (size_t)bh * HD * SS;   // [64][S]
  __bf16* pbuf = lds[wave];

  // Resident Q fragments for both head-dim chunks
  const v16bf aq0 = load_a(Qp, HD, q0 + nl, 0, hi);
  const v16bf aq1 = load_a(Qp, HD, q0 + nl, 32, hi);

  v8f o[4] = {vzero8(), vzero8(), vzero8(), vzero8()};
  float mrow[8], lrow[8];
#pragma unroll
  for (int r = 0; r < 8; ++r) { mrow[r] = -1e30f; lrow[r] = 0.0f; }

  const int jend = (q0 + 15) >> 5;   // inclusive causal key-tile bound

  // Preload K fragments for tile 0: [kchunk][key-half]
  v16bf kc[4];
  kc[0] = load_bt(Kp, HD, 0 + nl, 0, hi);
  kc[1] = load_bt(Kp, HD, 16 + nl, 0, hi);
  kc[2] = load_bt(Kp, HD, 0 + nl, 32, hi);
  kc[3] = load_bt(Kp, HD, 16 + nl, 32, hi);

#pragma unroll 2
  for (int j = 0; j <= jend; ++j) {
    const int key0 = j * 32;

    // V fragments for this tile, issued early (latency hidden by softmax VALU)
    v16bf bv[4];
#pragma unroll
    for (int t = 0; t < 4; ++t)
      bv[t] = *(const v16bf*)(Vp + (size_t)(t * 16 + nl) * SS + key0 + hi * 16);

    // scores 16x32 over HD=64
    v8f s0 = vzero8(), s1 = vzero8();
    s0 = wmma_bf16(aq0, kc[0], s0);
    s1 = wmma_bf16(aq0, kc[1], s1);
    s0 = wmma_bf16(aq1, kc[2], s0);
    s1 = wmma_bf16(aq1, kc[3], s1);

    // issue next tile's K fragment loads (in flight during softmax)
    const int keyn = (j < jend) ? key0 + 32 : key0;
    v16bf kn0 = load_bt(Kp, HD, keyn + nl, 0, hi);
    v16bf kn1 = load_bt(Kp, HD, keyn + 16 + nl, 0, hi);
    v16bf kn2 = load_bt(Kp, HD, keyn + nl, 32, hi);
    v16bf kn3 = load_bt(Kp, HD, keyn + 16 + nl, 32, hi);

    // causal mask (scale already folded into Q)
    if (key0 + 31 > q0) {
#pragma unroll
      for (int r = 0; r < 8; ++r) {
        const int rowg = q0 + r + 8 * hi;
        const int kk = key0 + nl;
        if (kk > rowg)      s0[r] = -1e30f;
        if (kk + 16 > rowg) s1[r] = -1e30f;
      }
    }

    // online softmax: row reductions across the 16 lanes holding one row
#pragma unroll
    for (int r = 0; r < 8; ++r) {
      float mx = fmaxf(s0[r], s1[r]);
#pragma unroll
      for (int off = 1; off < 16; off <<= 1) mx = fmaxf(mx, __shfl_xor(mx, off));
      const float mold = mrow[r];
      const float mnew = fmaxf(mold, mx);
      const float corr = __expf(mold - mnew);
      const float p0 = __expf(s0[r] - mnew);
      const float p1 = __expf(s1[r] - mnew);
      float rs = p0 + p1;
#pragma unroll
      for (int off = 1; off < 16; off <<= 1) rs += __shfl_xor(rs, off);
      lrow[r] = lrow[r] * corr + rs;
      mrow[r] = mnew;
#pragma unroll
      for (int t = 0; t < 4; ++t) o[t][r] *= corr;
      // stage P (C layout -> LDS [row][key] tile)
      pbuf[(r + 8 * hi) * 32 + nl]      = f2bf(p0);
      pbuf[(r + 8 * hi) * 32 + 16 + nl] = f2bf(p1);
    }

    // reload P as an A fragment (16x32): row = nl, K = hi*16 .. +15 (contiguous)
    AFrag pa;
    pa.h[0] = *(const v8bf*)(pbuf + nl * 32 + hi * 16);
    pa.h[1] = *(const v8bf*)(pbuf + nl * 32 + hi * 16 + 8);

    // o += P @ V
#pragma unroll
    for (int t = 0; t < 4; ++t) o[t] = wmma_bf16(pa.v, bv[t], o[t]);

    kc[0] = kn0; kc[1] = kn1; kc[2] = kn2; kc[3] = kn3;
  }

  // epilogue: divide by row sum, store bf16 into yb [B*S][D] (col = h*64 + d)
#pragma unroll
  for (int r = 0; r < 8; ++r) {
    const float inv = 1.0f / lrow[r];
    const int gm = b_ * SS + q0 + r + 8 * hi;
#pragma unroll
    for (int t = 0; t < 4; ++t) {
      yb[(size_t)gm * DD + h * HD + t * 16 + nl] = f2bf(o[t][r] * inv);
    }
  }
}

// ---------------- GEMM 2: out = yb @ Wo (f32 output) ----------------
// grid (M/256, D/64), block 256 (8 waves); wave tile 32x64, double-buffered + unrolled.
__global__ __launch_bounds__(256) void gemm_out_kernel(
    const __bf16* __restrict__ yb, const __bf16* __restrict__ woT,
    float* __restrict__ out) {
  const int lane = threadIdx.x & 31, wave = threadIdx.x >> 5;
  const int nl = lane & 15, hi = lane >> 4;
  const int m0 = blockIdx.x * 256 + wave * 32;
  const int n0 = blockIdx.y * 64;

  v8f c[2][4];
#pragma unroll
  for (int s = 0; s < 2; ++s)
#pragma unroll
    for (int t = 0; t < 4; ++t) c[s][t] = vzero8();

  v16bf a_c[2], b_c[4];
#pragma unroll
  for (int s = 0; s < 2; ++s) a_c[s] = load_a(yb, DD, m0 + s * 16 + nl, 0, hi);
#pragma unroll
  for (int t = 0; t < 4; ++t) b_c[t] = load_bt(woT, DD, n0 + t * 16 + nl, 0, hi);

#pragma unroll
  for (int k0 = 0; k0 < DD - 32; k0 += 32) {
    v16bf a_n[2], b_n[4];
#pragma unroll
    for (int s = 0; s < 2; ++s) a_n[s] = load_a(yb, DD, m0 + s * 16 + nl, k0 + 32, hi);
#pragma unroll
    for (int t = 0; t < 4; ++t) b_n[t] = load_bt(woT, DD, n0 + t * 16 + nl, k0 + 32, hi);

#pragma unroll
    for (int s = 0; s < 2; ++s)
#pragma unroll
      for (int t = 0; t < 4; ++t) c[s][t] = wmma_bf16(a_c[s], b_c[t], c[s][t]);

#pragma unroll
    for (int s = 0; s < 2; ++s) a_c[s] = a_n[s];
#pragma unroll
    for (int t = 0; t < 4; ++t) b_c[t] = b_n[t];
  }
#pragma unroll
  for (int s = 0; s < 2; ++s)
#pragma unroll
    for (int t = 0; t < 4; ++t) c[s][t] = wmma_bf16(a_c[s], b_c[t], c[s][t]);

#pragma unroll
  for (int s = 0; s < 2; ++s)
#pragma unroll
    for (int t = 0; t < 4; ++t) {
      const int n = n0 + t * 16 + nl;
#pragma unroll
      for (int r = 0; r < 8; ++r) {
        const int gm = m0 + s * 16 + r + 8 * hi;
        out[(size_t)gm * DD + n] = c[s][t][r];
      }
    }
}

// ---------------- Launch ----------------
extern "C" void kernel_launch(void* const* d_in, const int* in_sizes, int n_in,
                              void* d_out, int out_size, void* d_ws, size_t ws_size,
                              hipStream_t stream) {
  const float* x    = (const float*)d_in[0];   // [B,S,D]
  const float* Wqkv = (const float*)d_in[1];   // [D,3D]
  const float* Wo   = (const float*)d_in[2];   // [D,D]
  float* out = (float*)d_out;                  // [B,S,D]

  char* ws = (char*)d_ws;
  size_t off = 0;
  __bf16* wqkvT = (__bf16*)(ws + off); off += (size_t)N3 * DD * 2;   // 6 MB
  __bf16* woT   = (__bf16*)(ws + off); off += (size_t)DD * DD * 2;   // 2 MB
  __bf16* xb    = (__bf16*)(ws + off); off += (size_t)MM * DD * 2;   // 16 MB
  __bf16* Qb    = (__bf16*)(ws + off); off += (size_t)MM * DD * 2;   // 16 MB
  __bf16* Kb    = (__bf16*)(ws + off); off += (size_t)MM * DD * 2;   // 16 MB
  __bf16* Vb    = (__bf16*)(ws + off); off += (size_t)MM * DD * 2;   // 16 MB (transposed)
  __bf16* yb    = (__bf16*)(ws + off); off += (size_t)MM * DD * 2;   // 16 MB

  cvt_x_kernel   <<<(MM * DD) / 256, 256, 0, stream>>>(x, xb);
  cvt_wqkv_kernel<<<(DD * N3) / 256, 256, 0, stream>>>(Wqkv, wqkvT);
  cvt_wo_kernel  <<<(DD * DD) / 256, 256, 0, stream>>>(Wo, woT);

  gemm_qkv_kernel<<<dim3(MM / 256, N3 / 64), 256, 0, stream>>>(xb, wqkvT, Qb, Kb, Vb);

  attn_kernel<<<dim3(SS / 16 / 8, BB * HH), 256, 0, stream>>>(Qb, Kb, Vb, yb);

  gemm_out_kernel<<<dim3(MM / 256, DD / 64), 256, 0, stream>>>(yb, woT, out);
}
